// InvSGSSQuantizedLinear_64330020159906
// MI455X (gfx1250) — matrix operand order repaired and checked by reading, not compiled, not run on hardware
//
#include <hip/hip_runtime.h>
#include <hip/hip_bf16.h>
#include <stddef.h>
#include <stdint.h>

// ---------------------------------------------------------------------------
// InvSGSS quantized linear  ->  bf16 WMMA GEMM on gfx1250 (MI455X)
//   out[M,K] = x[M,N] @ W_deq[K,N]^T + bias,  M=8192, N=4096, K=4096
// Pre-pass dequantizes W (folding scale*mu2*mu1) and converts x to bf16;
// both (~100 MB) live in d_ws and fit MI455X's 192 MB L2.
// GEMM hot loop: global_load_async_to_lds_b128 (ASYNCcnt pipeline, no VGPR
// staging) + ds_load_b128 fragment gathers + v_wmma_f32_16x16x32_bf16.
// ---------------------------------------------------------------------------

typedef __attribute__((ext_vector_type(16))) __bf16 v16bf;
typedef __attribute__((ext_vector_type(8)))  float  v8f;
typedef __attribute__((ext_vector_type(4)))  unsigned int v4u;
typedef __attribute__((ext_vector_type(4)))  int   v4i;
typedef __attribute__((ext_vector_type(4)))  float v4f;

#define M_DIM 8192   // B*S
#define N_DIM 4096   // reduction
#define K_DIM 4096   // output features
#define BLK_M 128
#define BLK_K 128
#define BLK_N 64
#define LDS_STRIDE 72          // 64 + 8 pad: 144 B rows -> conflict-free b128

__device__ __forceinline__ unsigned short f2bf(float f) {
  unsigned int u = __float_as_uint(f);
  u += 0x7FFFu + ((u >> 16) & 1u);     // round-to-nearest-even
  return (unsigned short)(u >> 16);
}

union Frag { v16bf v; v4u u[2]; };

// -------------------- pre-pass 1: x (fp32) -> bf16 -------------------------
__global__ void __launch_bounds__(256)
convert_x_kernel(const float* __restrict__ x, unsigned short* __restrict__ xb) {
  size_t t = (size_t)blockIdx.x * 256 + threadIdx.x;   // one 8-elem chunk
  const v4f* p = (const v4f*)x + t * 2;
  v4f f0 = p[0], f1 = p[1];
  union { unsigned short s16[8]; v4u u; } r;
#pragma unroll
  for (int j = 0; j < 4; ++j) {
    r.s16[j]     = f2bf(f0[j]);
    r.s16[j + 4] = f2bf(f1[j]);
  }
  ((v4u*)xb)[t] = r.u;
}

// ------ pre-pass 2: dequant W_q -> bf16, fold scale*mu2*mu1 ---------------
__global__ void __launch_bounds__(256)
dequant_w_kernel(const int* __restrict__ Wq, const float* __restrict__ scales,
                 const float* __restrict__ zeros, const float* __restrict__ mu1,
                 const float* __restrict__ mu2, unsigned short* __restrict__ wb) {
  int t  = blockIdx.x * 256 + threadIdx.x;   // chunk of 8 along n
  int k  = t >> 9;                            // N/8 = 512 chunks per row
  int nc = (t & 511) << 3;
  int g  = nc >> 7;                           // GROUP_SIZE = 128
  float s = scales[k * 32 + g] * mu2[k];
  float z = zeros[k * 32 + g];
  const v4i* q = (const v4i*)(Wq + (size_t)k * N_DIM + nc);
  const v4f* m = (const v4f*)(mu1 + nc);
  v4i q0 = q[0], q1 = q[1];
  v4f m0 = m[0], m1 = m[1];
  union { unsigned short s16[8]; v4u u; } r;
#pragma unroll
  for (int j = 0; j < 4; ++j) {
    r.s16[j]     = f2bf(((float)q0[j] - z) * s * m0[j]);
    r.s16[j + 4] = f2bf(((float)q1[j] - z) * s * m1[j]);
  }
  *(v4u*)(wb + (size_t)k * N_DIM + nc) = r.u;
}

// -------------------- main bf16 WMMA GEMM ---------------------------------
// one b128 async copy: LDS[ldsoff + OFF] = MEM[s_base + vaddr + OFF]
#define ASYNC_B128(ldsoff, vaddr, sbase, OFF)                                  \
  asm volatile("global_load_async_to_lds_b128 %0, %1, %2 offset:" #OFF         \
               :: "v"(ldsoff), "v"(vaddr), "s"(sbase) : "memory")

__global__ void __launch_bounds__(256)
gemm_bf16_wmma(const unsigned short* __restrict__ xb,
               const unsigned short* __restrict__ wb,
               const float* __restrict__ bias,
               float* __restrict__ out) {
  __shared__ __align__(16) unsigned short shX[2][BLK_M * LDS_STRIDE];
  __shared__ __align__(16) unsigned short shW[2][BLK_K * LDS_STRIDE];

  const int tid  = threadIdx.x;
  const int lane = tid & 31;
  const int wave = tid >> 5;
  const int lo   = lane & 15;
  const int hi   = lane >> 4;

  const int m0 = blockIdx.y * BLK_M;
  const int k0 = blockIdx.x * BLK_K;      // K-tile fastest -> L2 reuse of x rows

  const int wm = (wave & 1) * 64;         // wave's m offset (2 waves over M)
  const int wk = (wave >> 1) * 32;        // wave's k offset (4 waves over K)

  // staging: each thread moves 64 B of X and of W per tile (4x async b128)
  const int srow = tid >> 1;              // 0..127
  const int scol = (tid & 1) * 32;        // 0 / 32

  // global byte offsets (32-bit: buffers are < 128 MB), SGPR64 base + VGPR off
  const uint32_t goffX = ((uint32_t)(m0 + srow) * N_DIM + (uint32_t)scol) * 2u;
  const uint32_t goffW = ((uint32_t)(k0 + srow) * N_DIM + (uint32_t)scol) * 2u;
  const unsigned long long baseX = (unsigned long long)(uintptr_t)xb;
  const unsigned long long baseW = (unsigned long long)(uintptr_t)wb;

  // LDS byte offsets: low 32 bits of a generic LDS pointer are the LDS offset
  const uint32_t lrow  = ((uint32_t)srow * LDS_STRIDE + (uint32_t)scol) * 2u;
  const uint32_t ldsX0 = (uint32_t)(uintptr_t)(&shX[0][0]) + lrow;
  const uint32_t ldsX1 = (uint32_t)(uintptr_t)(&shX[1][0]) + lrow;
  const uint32_t ldsW0 = (uint32_t)(uintptr_t)(&shW[0][0]) + lrow;
  const uint32_t ldsW1 = (uint32_t)(uintptr_t)(&shW[1][0]) + lrow;

  auto issue_tile = [&](int n0, int buf) {
    const uint32_t vx = goffX + (uint32_t)(n0 * 2);
    const uint32_t vw = goffW + (uint32_t)(n0 * 2);
    const uint32_t lx = buf ? ldsX1 : ldsX0;
    const uint32_t lw = buf ? ldsW1 : ldsW0;
    ASYNC_B128(lx, vx, baseX, 0);
    ASYNC_B128(lx, vx, baseX, 16);
    ASYNC_B128(lx, vx, baseX, 32);
    ASYNC_B128(lx, vx, baseX, 48);
    ASYNC_B128(lw, vw, baseW, 0);
    ASYNC_B128(lw, vw, baseW, 16);
    ASYNC_B128(lw, vw, baseW, 32);
    ASYNC_B128(lw, vw, baseW, 48);
  };

  v8f acc[4][2];
  const v8f vzero = {0.f, 0.f, 0.f, 0.f, 0.f, 0.f, 0.f, 0.f};
#pragma unroll
  for (int mf = 0; mf < 4; ++mf)
#pragma unroll
    for (int kf = 0; kf < 2; ++kf) acc[mf][kf] = vzero;

  auto compute = [&](int buf) {
    const unsigned short* xs = &shX[buf][0];
    const unsigned short* ws = &shW[buf][0];
#pragma unroll
    for (int c = 0; c < 2; ++c) {          // two 16x16x32 K-steps per tile
      Frag a[4], b[2];
#pragma unroll
      for (int mf = 0; mf < 4; ++mf) {
        // A 16x32 bf16: lane = row (lo); chunks at K = c*32 + hi*8, +16
        const unsigned short* p =
            xs + (wm + mf * 16 + lo) * LDS_STRIDE + c * 32 + hi * 8;
        a[mf].u[0] = *(const v4u*)p;
        a[mf].u[1] = *(const v4u*)(p + 16);
      }
#pragma unroll
      for (int kf = 0; kf < 2; ++kf) {
        // B 32x16 bf16: lane = column (lo) = W row; chunks at K = c*32 + hi*8, +16
        const unsigned short* p =
            ws + (wk + kf * 16 + lo) * LDS_STRIDE + c * 32 + hi * 8;
        b[kf].u[0] = *(const v4u*)p;
        b[kf].u[1] = *(const v4u*)(p + 16);
      }
#pragma unroll
      for (int mf = 0; mf < 4; ++mf)
#pragma unroll
        for (int kf = 0; kf < 2; ++kf)
          acc[mf][kf] = __builtin_amdgcn_wmma_f32_16x16x32_bf16(
              false, a[mf].v, false, b[kf].v, (short)0, acc[mf][kf],
              false, false);
    }
  };

  const int NT = N_DIM / BLK_N;   // 64 reduction tiles
  issue_tile(0, 0);
#pragma unroll 1
  for (int it = 0; it < NT; ++it) {
    const int cur = it & 1;
    if (it + 1 < NT) {
      issue_tile((it + 1) * BLK_N, cur ^ 1);
      // 8 newest (tile it+1) may remain in flight; oldest 8 (tile it) done.
      asm volatile("s_wait_asynccnt 0x8" ::: "memory");
    } else {
      asm volatile("s_wait_asynccnt 0x0" ::: "memory");
    }
    __syncthreads();             // all waves' tile-it data visible in LDS
    compute(cur);
    __syncthreads();             // all waves done reading buf `cur`
  }

  // epilogue: C/D layout -> VGPR j: row j (lanes 0-15) / row j+8 (lanes 16-31)
#pragma unroll
  for (int kf = 0; kf < 2; ++kf) {
    const int col = k0 + wk + kf * 16 + lo;
    const float bv = bias[col];
#pragma unroll
    for (int mf = 0; mf < 4; ++mf) {
      const int rbase = m0 + wm + mf * 16 + hi * 8;
#pragma unroll
      for (int j = 0; j < 8; ++j)
        out[(size_t)(rbase + j) * K_DIM + col] = acc[mf][kf][j] + bv;
    }
  }
}

// ---------------------------------------------------------------------------
extern "C" void kernel_launch(void* const* d_in, const int* in_sizes, int n_in,
                              void* d_out, int out_size, void* d_ws, size_t ws_size,
                              hipStream_t stream) {
  const float* x      = (const float*)d_in[0];
  const int*   Wq     = (const int*)d_in[1];
  const float* scales = (const float*)d_in[2];
  const float* zeros  = (const float*)d_in[3];
  const float* mu1    = (const float*)d_in[4];
  const float* mu2    = (const float*)d_in[5];
  const float* bias   = (const float*)d_in[6];
  float* out = (float*)d_out;

  // workspace layout: xb (bf16, M*N) then wb (bf16, K*N)  ~100.7 MB total
  unsigned short* xbuf = (unsigned short*)d_ws;
  unsigned short* wbuf = xbuf + (size_t)M_DIM * N_DIM;

  convert_x_kernel<<<(M_DIM * (N_DIM / 8)) / 256, 256, 0, stream>>>(x, xbuf);
  dequant_w_kernel<<<(K_DIM * (N_DIM / 8)) / 256, 256, 0, stream>>>(
      Wq, scales, zeros, mu1, mu2, wbuf);

  dim3 grid(K_DIM / BLK_K, M_DIM / BLK_M);   // (32, 64), K-tile fastest
  gemm_bf16_wmma<<<grid, 256, 0, stream>>>(xbuf, wbuf, bias, out);
}
